// GCLMCDR_53326313947268
// MI455X (gfx1250) — matrix operand order repaired and checked by reading, not compiled, not run on hardware
//
#include <hip/hip_runtime.h>

// ---------------------------------------------------------------------------
// Types for CDNA5 WMMA (wave32, 16x16x32 f16 -> f32)
// ---------------------------------------------------------------------------
typedef _Float16 f16;
typedef __attribute__((ext_vector_type(2)))  float    f32x2;
typedef __attribute__((ext_vector_type(8)))  _Float16 f16x8;
typedef __attribute__((ext_vector_type(16))) _Float16 f16x16;
typedef __attribute__((ext_vector_type(8)))  float    f32x8;

union AFrag { f16x16 v; f16x8 h[2]; };
union Pack16 { f16x8 v8[2]; f16 s[16]; };

#define LDT 40  // LDS row stride in halves for a 32-wide K tile (+8 pad)

static __host__ __device__ inline int roundup32(int k) { return (k + 31) & ~31; }

// ---------------------------------------------------------------------------
// Weight prep: fp32 B(KxN) -> f16 Bt(N x Kp), transposed, zero-padded to Kp.
// ---------------------------------------------------------------------------
__global__ void cvt_tr_k(const float* __restrict__ B, f16* __restrict__ Bt,
                         int K, int N, int Kp) {
  long idx = (long)blockIdx.x * blockDim.x + threadIdx.x;
  if (idx >= (long)N * Kp) return;
  int n = (int)(idx / Kp), k = (int)(idx % Kp);
  Bt[idx] = (k < K) ? (f16)B[(long)k * N + n] : (f16)0.f;
}

// ---------------------------------------------------------------------------
// Tiled WMMA GEMM: C(MxN) = A(MxK) @ B(KxN); A fp32 (converted in LDS),
// B pre-transposed f16 (Bt), C fp32. Fused bias / eval-BN / ReLU epilogue.
// Block = 256 threads (8 wave32), 128x128 tile; wave -> 32x64 subtile.
// M edge: A rows clamped to M-1 (row-independent GEMM; OOB stores dropped).
// K tail: branch-free clamp + select-to-zero staging.
// ---------------------------------------------------------------------------
__global__ __launch_bounds__(256) void wmma_gemm_f16(
    const float* __restrict__ A, const f16* __restrict__ Bt,
    float* __restrict__ C, int M, int N, int K, int Kp,
    const float* __restrict__ bias, const float* __restrict__ bn_g,
    const float* __restrict__ bn_be, int relu)
{
  __shared__ _Float16 As[128 * LDT];
  __shared__ _Float16 Bs[128 * LDT];
  const int t    = threadIdx.x;
  const int lane = t & 31;
  const int wid  = t >> 5;
  const int wr   = wid & 3;   // 4 wave rows of 32
  const int wc   = wid >> 2;  // 2 wave cols of 64
  const long mbase = (long)blockIdx.x * 128;
  const long nbase = (long)blockIdx.y * 128;
  const bool fullM = (mbase + 128) <= M;

  f32x8 acc[2][4];
#pragma unroll
  for (int mi = 0; mi < 2; ++mi)
#pragma unroll
    for (int ni = 0; ni < 4; ++ni)
#pragma unroll
      for (int i = 0; i < 8; ++i) acc[mi][ni][i] = 0.f;

  const int arow = t >> 1;        // 0..127
  const int aks  = (t & 1) << 4;  // 0 | 16
  // clamp row: C rows are independent, OOB rows never stored in epilogue
  const long growc = (mbase + arow < M) ? (mbase + arow) : (M - 1);
  const float* aprow = A + growc * (long)K;
  const f16*  brow  = Bt + (nbase + arow) * (long)Kp + aks;

  for (int kt = 0; kt < K; kt += 32) {
    __syncthreads();
    // ---- stage A tile (128 x 32) fp32 -> f16 ----
    if ((kt + 32) <= K) {  // full K slab: vectorized (8B aligned; K even everywhere)
      const float* ap = aprow + kt + aks;
      Pack16 u;
#pragma unroll
      for (int j = 0; j < 8; ++j) {
        f32x2 f = *(const f32x2*)(ap + 2 * j);
        u.s[2 * j]     = (f16)f.x;
        u.s[2 * j + 1] = (f16)f.y;
      }
      *(f16x8*)&As[arow * LDT + aks]     = u.v8[0];
      *(f16x8*)&As[arow * LDT + aks + 8] = u.v8[1];
      if ((kt + 32) < K)
        __builtin_prefetch(ap + 32, 0, 1);  // global_prefetch_b8 next K slab
    } else {               // K tail: branch-free clamp + select to zero
      Pack16 u;
#pragma unroll
      for (int j = 0; j < 16; ++j) {
        const int k  = kt + aks + j;
        const int kk = (k < K) ? k : (K - 1);
        float v = aprow[kk];
        v = (k < K) ? v : 0.f;
        u.s[j] = (f16)v;
      }
      *(f16x8*)&As[arow * LDT + aks]     = u.v8[0];
      *(f16x8*)&As[arow * LDT + aks + 8] = u.v8[1];
    }
    // ---- stage B tile from pre-transposed f16 (always in-bounds: Kp pad) ----
    {
      const f16* bp = brow + kt;
      *(f16x8*)&Bs[arow * LDT + aks]     = *(const f16x8*)bp;
      *(f16x8*)&Bs[arow * LDT + aks + 8] = *(const f16x8*)(bp + 8);
    }
    __syncthreads();

    const int mlane = lane & 15;
    const int akb = (lane < 16) ? 0 : 8;   // A frag K-base per ISA layout
    const int bkb = (lane < 16) ? 0 : 8;   // B frag K-base per ISA layout
    AFrag a[2], b[4];
#pragma unroll
    for (int mi = 0; mi < 2; ++mi) {
      const int r = wr * 32 + mi * 16 + mlane;
      a[mi].h[0] = *(const f16x8*)&As[r * LDT + akb];
      a[mi].h[1] = *(const f16x8*)&As[r * LDT + 16 + akb];
    }
#pragma unroll
    for (int ni = 0; ni < 4; ++ni) {
      const int r = wc * 64 + ni * 16 + mlane;
      b[ni].h[0] = *(const f16x8*)&Bs[r * LDT + bkb];
      b[ni].h[1] = *(const f16x8*)&Bs[r * LDT + 16 + bkb];
    }
#pragma unroll
    for (int mi = 0; mi < 2; ++mi)
#pragma unroll
      for (int ni = 0; ni < 4; ++ni)
        acc[mi][ni] = __builtin_amdgcn_wmma_f32_16x16x32_f16(
            false, a[mi].v, false, b[ni].v, (short)0, acc[mi][ni], false, false);
  }

  // ---- epilogue (n < N always: grid.y == N/128) ----
  const float bnscale = rsqrtf(1.f + 1e-5f);
#pragma unroll
  for (int mi = 0; mi < 2; ++mi) {
#pragma unroll
    for (int ni = 0; ni < 4; ++ni) {
      const long n = nbase + wc * 64 + ni * 16 + (lane & 15);
      float bv = bias ? bias[n] : 0.f;
      float sv = bn_g ? bn_g[n] * bnscale : 1.f;
      float ev = bn_be ? bn_be[n] : 0.f;
#pragma unroll
      for (int i = 0; i < 8; ++i) {
        const long m = mbase + wr * 32 + mi * 16 + i + ((lane < 16) ? 0 : 8);
        if (fullM || m < M) {
          float v = acc[mi][ni][i] + bv;
          v = v * sv + ev;
          if (relu) v = fmaxf(v, 0.f);
          C[m * (long)N + n] = v;
        }
      }
    }
  }
}

// ---------------------------------------------------------------------------
// Small elementwise / graph kernels (deterministic, no float atomics)
// ---------------------------------------------------------------------------
__global__ void mulcat_k(const float* __restrict__ MG, const float* __restrict__ ME,
                         const float* __restrict__ MN, float* __restrict__ MC,
                         int Bn, int D) {
  long idx = (long)blockIdx.x * blockDim.x + threadIdx.x;
  long tot = (long)Bn * 3 * D;
  if (idx >= tot) return;
  int b = (int)(idx / (3 * D)); int c = (int)(idx % (3 * D));
  int seg = c / D, d = c % D;
  float g = MG[(long)b * D + d], e = ME[(long)b * D + d], n = MN[(long)b * D + d];
  MC[idx] = (seg == 0) ? g * e : (seg == 1) ? g * n : e * n;
}

__global__ void gat_coeff_k(const float* __restrict__ hf, const float* __restrict__ wsrc,
                            const float* __restrict__ wdst, float* __restrict__ asrc,
                            float* __restrict__ adst, int Nn, int heads, int D) {
  int idx = blockIdx.x * blockDim.x + threadIdx.x;
  if (idx >= Nn * heads) return;
  int n = idx / heads, h = idx % heads;
  const float* row = hf + (long)n * heads * D + (long)h * D;
  float sa = 0.f, sd = 0.f;
  for (int d = 0; d < D; ++d) { float v = row[d]; sa += v * wsrc[h * D + d]; sd += v * wdst[h * D + d]; }
  asrc[idx] = sa; adst[idx] = sd;
}

__device__ __forceinline__ float lrelu02(float x) { return x >= 0.f ? x : 0.2f * x; }

__global__ void gat_weights_k(const float* __restrict__ asrc, const float* __restrict__ adst,
                              float* __restrict__ wts, int Nn, int heads, int npg) {
  int idx = blockIdx.x * blockDim.x + threadIdx.x;
  if (idx >= Nn * heads) return;
  int n = idx / heads, h = idx % heads;
  float ad = adst[idx];
  bool hp = (n % npg) != 0;
  bool hn = (n % npg) != npg - 1;
  float a0 = lrelu02(asrc[idx] + ad);
  float a1 = hp ? lrelu02(asrc[(n - 1) * heads + h] + ad) : -1e30f;
  float a2 = hn ? lrelu02(asrc[(n + 1) * heads + h] + ad) : -1e30f;
  float m  = fmaxf(a0, fmaxf(a1, a2));
  float e0 = __expf(a0 - m);
  float e1 = hp ? __expf(a1 - m) : 0.f;
  float e2 = hn ? __expf(a2 - m) : 0.f;
  float inv = 1.f / (e0 + e1 + e2);
  wts[(long)idx * 3 + 0] = e0 * inv;
  wts[(long)idx * 3 + 1] = e1 * inv;
  wts[(long)idx * 3 + 2] = e2 * inv;
}

__global__ void gat_agg_k(const float* __restrict__ hf, const float* __restrict__ wts,
                          const float* __restrict__ bias, const float* __restrict__ g,
                          const float* __restrict__ be, float* __restrict__ out,
                          int Nn, int heads, int D, int npg) {
  long idx = (long)blockIdx.x * blockDim.x + threadIdx.x;
  long tot = (long)Nn * heads * D;
  if (idx >= tot) return;
  int hd = heads * D;
  int n = (int)(idx / hd), c = (int)(idx % hd), h = c / D;
  const float* w = wts + ((long)n * heads + h) * 3;
  float v = w[0] * hf[idx];
  if (n % npg)              v += w[1] * hf[idx - hd];
  if ((n % npg) != npg - 1) v += w[2] * hf[idx + hd];
  v += bias[c];
  v = v * (g[c] * rsqrtf(1.f + 1e-5f)) + be[c];
  out[idx] = fmaxf(v, 0.f);
}

__global__ void maxpool_k(const float* __restrict__ x, float* __restrict__ hD,
                          int Bn, int npg, int D) {
  long idx = (long)blockIdx.x * blockDim.x + threadIdx.x;
  if (idx >= (long)Bn * D) return;
  int b = (int)(idx / D), d = (int)(idx % D);
  float m = -1e30f;
  for (int a = 0; a < npg; ++a) m = fmaxf(m, x[((long)b * npg + a) * D + d]);
  hD[idx] = m;
}

__global__ void node_stats_k(const int* __restrict__ cells, const int* __restrict__ drugs,
                             int* __restrict__ lastb, float* __restrict__ dis,
                             int total, int ncells, int Bn) {
  int t = blockIdx.x * blockDim.x + threadIdx.x;
  if (t >= total) return;
  int last = -1, cnt = 1;  // self loop
  if (t < ncells) { for (int p = 0; p < Bn; ++p) if (cells[p] == t) { last = p; ++cnt; } }
  else { int dd = t - ncells; for (int p = 0; p < Bn; ++p) if (drugs[p] == dd) { last = p; ++cnt; } }
  lastb[t] = last;
  dis[t] = rsqrtf((float)cnt);
}

__global__ void build_h0_k(const float* __restrict__ proj, const float* __restrict__ hD,
                           const int* __restrict__ lastb, float* __restrict__ h0,
                           int total, int ncells, int D) {
  long idx = (long)blockIdx.x * blockDim.x + threadIdx.x;
  if (idx >= (long)total * D) return;
  int t = (int)(idx / D), d = (int)(idx % D);
  int lb = lastb[t];
  float v = 0.f;
  if (lb >= 0) v = (t < ncells) ? proj[(long)lb * D + d] : hD[(long)lb * D + d];
  h0[idx] = v;
}

__global__ void gcn_agg_k(const float* __restrict__ hW, const float* __restrict__ dis,
                          const int* __restrict__ cells, const int* __restrict__ drugs,
                          const float* __restrict__ b, const float* __restrict__ prelu_a,
                          float* __restrict__ h1, int total, int ncells, int Bn, int D) {
  long idx = (long)blockIdx.x * blockDim.x + threadIdx.x;
  if (idx >= (long)total * D) return;
  int t = (int)(idx / D), d = (int)(idx % D);
  float dt = dis[t];
  float acc = dt * dt * hW[idx];  // self loop
  if (t < ncells) {
    for (int p = 0; p < Bn; ++p)
      if (cells[p] == t) { int s = ncells + drugs[p]; acc += dt * dis[s] * hW[(long)s * D + d]; }
  } else {
    int dd = t - ncells;
    for (int p = 0; p < Bn; ++p)
      if (drugs[p] == dd) { int s = cells[p]; acc += dt * dis[s] * hW[(long)s * D + d]; }
  }
  acc += b[d];
  float a = prelu_a[0];
  h1[idx] = acc >= 0.f ? acc : a * acc;
}

__global__ void concat2_k(const float* __restrict__ A, const float* __restrict__ B,
                          float* __restrict__ C, int n, int D) {
  long idx = (long)blockIdx.x * blockDim.x + threadIdx.x;
  if (idx >= (long)n * 2 * D) return;
  int t = (int)(idx / (2 * D)), c = (int)(idx % (2 * D));
  C[idx] = (c < D) ? A[(long)t * D + c] : B[(long)t * D + (c - D)];
}

__global__ void dot_rows_k(const float* __restrict__ X, const float* __restrict__ w,
                           const float* __restrict__ b, float* __restrict__ out,
                           int n, int D, int sigm) {
  int t = blockIdx.x * blockDim.x + threadIdx.x;
  if (t >= n) return;
  float s = 0.f;
  for (int d = 0; d < D; ++d) s += X[(long)t * D + d] * w[d];
  if (b) s += b[0];
  out[t] = sigm ? 1.f / (1.f + __expf(-s)) : s;
}

__global__ void softmax_k(const float* __restrict__ scores, float* __restrict__ attn, int n) {
  __shared__ float red[256];
  int t = threadIdx.x;
  float m = -1e30f;
  for (int i = t; i < n; i += 256) m = fmaxf(m, scores[i]);
  red[t] = m; __syncthreads();
  for (int s = 128; s > 0; s >>= 1) { if (t < s) red[t] = fmaxf(red[t], red[t + s]); __syncthreads(); }
  m = red[0]; __syncthreads();
  float sum = 0.f;
  for (int i = t; i < n; i += 256) sum += __expf(scores[i] - m);
  red[t] = sum; __syncthreads();
  for (int s = 128; s > 0; s >>= 1) { if (t < s) red[t] += red[t + s]; __syncthreads(); }
  float inv = 1.f / red[0];
  for (int i = t; i < n; i += 256) attn[i] = __expf(scores[i] - m) * inv;
}

__global__ void weighted_sum_k(const float* __restrict__ h1, const float* __restrict__ attn,
                               float* __restrict__ tod, int n, int D) {
  int d = blockIdx.x * blockDim.x + threadIdx.x;
  if (d >= D) return;
  float s = 0.f;
  for (int t = 0; t < n; ++t) s += attn[t] * h1[(long)t * D + d];
  tod[d] = s;
}

__global__ void pred_k(const float* __restrict__ h1, const int* __restrict__ cells,
                       const int* __restrict__ drugs, const float* __restrict__ w,
                       const float* __restrict__ b, float* __restrict__ out,
                       int Bn, int ncells, int D) {
  int p = blockIdx.x * blockDim.x + threadIdx.x;
  if (p >= Bn) return;
  const float* hc = h1 + (long)cells[p] * D;
  const float* hd = h1 + (long)(ncells + drugs[p]) * D;
  float s = 0.f;
  for (int d = 0; d < D; ++d) s += hc[d] * hd[d] * w[d];
  s += b[0];
  out[p] = 1.f / (1.f + __expf(-s));
}

// ---------------------------------------------------------------------------
// Host launcher
// ---------------------------------------------------------------------------
static inline int cdiv_i(long a, int b) { return (int)((a + b - 1) / b); }

extern "C" void kernel_launch(void* const* d_in, const int* in_sizes, int n_in,
                              void* d_out, int out_size, void* d_ws, size_t ws_size,
                              hipStream_t stream) {
  (void)in_sizes; (void)n_in; (void)out_size; (void)ws_size;
  constexpr int H = 128, HEADS = 2, Bn = 1024, NPG = 40, NDN = Bn * NPG, ATOM = 78;
  constexpr int GDIM = 20000, EDIM = 15000, TDIM = 17000;
  constexpr int NCELLS = 500, TOTAL = 700;

  auto F = [&](int i) { return (const float*)d_in[i]; };
  // params flattened leaf-by-leaf in sorted-key pytree order
  const float *BN1_B = F(0), *BN1_G = F(1), *BN2_B = F(2), *BN2_G = F(3);
  const float *DISCW = F(4);
  const float *EPI_W1 = F(5), *EPI_W2 = F(6), *EPI_W3 = F(7), *EPI_B1 = F(8), *EPI_B2 = F(9),
              *EPI_B3 = F(10), *EPI_BE1 = F(11), *EPI_BE2 = F(12), *EPI_G1 = F(13), *EPI_G2 = F(14);
  const float *G1_W = F(15), *G1_ADST = F(16), *G1_ASRC = F(17), *G1_B = F(18);
  const float *G2_W = F(19), *G2_ADST = F(20), *G2_ASRC = F(21), *G2_B = F(22);
  const float *GCN_W = F(23), *GCN_B = F(24);
  const float *GEN_W1 = F(25), *GEN_W2 = F(26), *GEN_W3 = F(27), *GEN_B1 = F(28), *GEN_B2 = F(29),
              *GEN_B3 = F(30), *GEN_BE1 = F(31), *GEN_BE2 = F(32), *GEN_G1 = F(33), *GEN_G2 = F(34);
  const float *PRED_W = F(35), *PRED_B = F(36), *PRELU = F(37);
  const float *PROJ_W = F(38), *PROJ_B = F(39);
  const float *RD_W1 = F(40), *RD_W2 = F(41), *RD_B1 = F(42), *RD_B2 = F(43);
  const float *TRA_W1 = F(44), *TRA_W2 = F(45), *TRA_W3 = F(46), *TRA_B1 = F(47), *TRA_B2 = F(48),
              *TRA_B3 = F(49), *TRA_BE1 = F(50), *TRA_BE2 = F(51), *TRA_G1 = F(52), *TRA_G2 = F(53);
  const float *DRUG_X = F(54);
  (void)d_in[55]; (void)d_in[56];  // edge_index / batch: chain structure is implicit
  const float *GENO = F(57), *EPIG = F(58), *TRAN = F(59);
  const int *CELLS = (const int*)d_in[60], *DRUGS = (const int*)d_in[61];
  (void)d_in[62];  // labels unused in forward

  // workspace layout (float offsets; all 64-float aligned)
  float* W = (float*)d_ws;
  constexpr size_t O_T1 = 0;                         // 1024*256
  constexpr size_t O_T2 = O_T1 + 262144;             // 1024*128
  constexpr size_t O_MG = O_T2 + 131072;
  constexpr size_t O_ME = O_MG + 131072;
  constexpr size_t O_MN = O_ME + 131072;
  constexpr size_t O_MC = O_MN + 131072;             // 1024*384
  constexpr size_t O_PROJ = O_MC + 393216;
  constexpr size_t O_HF = O_PROJ + 131072;           // 40960*256 (also hfeat2)
  constexpr size_t O_X1 = O_HF + 10485760;           // 40960*256 (also x2)
  constexpr size_t O_ASRC = O_X1 + 10485760;
  constexpr size_t O_ADST = O_ASRC + 81920;
  constexpr size_t O_WTS = O_ADST + 81920;           // 40960*2*3
  constexpr size_t O_HD = O_WTS + 245760;            // 1024*128
  constexpr size_t O_H0 = O_HD + 131072;             // 700*128
  constexpr size_t O_HW = O_H0 + 89600;
  constexpr size_t O_H1 = O_HW + 89600;
  constexpr size_t O_CAT = O_H1 + 89600;             // 700*256
  constexpr size_t O_SB = O_CAT + 179200;
  constexpr size_t O_SC = O_SB + 89600;              // 700 scores
  constexpr size_t O_LAST = O_SC + 704;              // 700 ints
  constexpr size_t O_DIS = O_LAST + 704;
  constexpr size_t O_V = O_DIS + 704;                // 128
  constexpr size_t O_BT = O_V + 192;                 // shared f16 Bt scratch: 256*20000 f16
  f16* BT = (f16*)(W + O_BT);

  float* out = (float*)d_out;
  float* OUT_PRED = out;            // [0,1024)
  float* OUT_TOD  = out + 1024;     // [1024,1152)
  float* OUT_ATTN = out + 1152;     // [1152,1852)
  float* OUT_DISC = out + 1852;     // [1852,2552)

  auto gemm = [&](float* C, const float* A, const float* Bm, int M, int N, int K,
                  const float* bias, const float* g, const float* be, int relu) {
    const int Kp = roundup32(K);
    hipLaunchKernelGGL(cvt_tr_k, dim3((unsigned)cdiv_i((long)N * Kp, 256)), dim3(256), 0, stream,
                       Bm, BT, K, N, Kp);
    dim3 grid((unsigned)cdiv_i(M, 128), (unsigned)(N / 128));
    hipLaunchKernelGGL(wmma_gemm_f16, grid, dim3(256), 0, stream,
                       A, BT, C, M, N, K, Kp, bias, g, be, relu);
  };
  auto ew = [&](long tot) { return dim3((unsigned)cdiv_i(tot, 256)); };

  // ---- omics DNNs: Linear-BN-ReLU x2 -> Linear ----
  gemm(W + O_T1, GENO, GEN_W1, Bn, 256, GDIM, GEN_B1, GEN_G1, GEN_BE1, 1);
  gemm(W + O_T2, W + O_T1, GEN_W2, Bn, 128, 256, GEN_B2, GEN_G2, GEN_BE2, 1);
  gemm(W + O_MG, W + O_T2, GEN_W3, Bn, 128, 128, GEN_B3, nullptr, nullptr, 0);
  gemm(W + O_T1, EPIG, EPI_W1, Bn, 256, EDIM, EPI_B1, EPI_G1, EPI_BE1, 1);
  gemm(W + O_T2, W + O_T1, EPI_W2, Bn, 128, 256, EPI_B2, EPI_G2, EPI_BE2, 1);
  gemm(W + O_ME, W + O_T2, EPI_W3, Bn, 128, 128, EPI_B3, nullptr, nullptr, 0);
  gemm(W + O_T1, TRAN, TRA_W1, Bn, 256, TDIM, TRA_B1, TRA_G1, TRA_BE1, 1);
  gemm(W + O_T2, W + O_T1, TRA_W2, Bn, 128, 256, TRA_B2, TRA_G2, TRA_BE2, 1);
  gemm(W + O_MN, W + O_T2, TRA_W3, Bn, 128, 128, TRA_B3, nullptr, nullptr, 0);
  hipLaunchKernelGGL(mulcat_k, ew((long)Bn * 3 * H), dim3(256), 0, stream,
                     W + O_MG, W + O_ME, W + O_MN, W + O_MC, Bn, H);

  // ---- GAT layer 1 (heads=2, concat) + BN + ReLU ----
  gemm(W + O_HF, DRUG_X, G1_W, NDN, HEADS * H, ATOM, nullptr, nullptr, nullptr, 0);
  hipLaunchKernelGGL(gat_coeff_k, ew((long)NDN * HEADS), dim3(256), 0, stream,
                     W + O_HF, G1_ASRC, G1_ADST, W + O_ASRC, W + O_ADST, NDN, HEADS, H);
  hipLaunchKernelGGL(gat_weights_k, ew((long)NDN * HEADS), dim3(256), 0, stream,
                     W + O_ASRC, W + O_ADST, W + O_WTS, NDN, HEADS, NPG);
  hipLaunchKernelGGL(gat_agg_k, ew((long)NDN * HEADS * H), dim3(256), 0, stream,
                     W + O_HF, W + O_WTS, G1_B, BN1_G, BN1_B, W + O_X1, NDN, HEADS, H, NPG);

  // ---- GAT layer 2 (heads=1) + BN + ReLU ----
  gemm(W + O_HF, W + O_X1, G2_W, NDN, H, HEADS * H, nullptr, nullptr, nullptr, 0);
  hipLaunchKernelGGL(gat_coeff_k, ew((long)NDN), dim3(256), 0, stream,
                     W + O_HF, G2_ASRC, G2_ADST, W + O_ASRC, W + O_ADST, NDN, 1, H);
  hipLaunchKernelGGL(gat_weights_k, ew((long)NDN), dim3(256), 0, stream,
                     W + O_ASRC, W + O_ADST, W + O_WTS, NDN, 1, NPG);
  hipLaunchKernelGGL(gat_agg_k, ew((long)NDN * H), dim3(256), 0, stream,
                     W + O_HF, W + O_WTS, G2_B, BN2_G, BN2_B, W + O_X1, NDN, 1, H, NPG);
  hipLaunchKernelGGL(maxpool_k, ew((long)Bn * H), dim3(256), 0, stream,
                     W + O_X1, W + O_HD, Bn, NPG, H);

  // ---- bipartite graph, 1-layer GCN + PReLU ----
  gemm(W + O_PROJ, W + O_MC, PROJ_W, Bn, 128, 3 * H, PROJ_B, nullptr, nullptr, 0);
  hipLaunchKernelGGL(node_stats_k, ew((long)TOTAL), dim3(256), 0, stream,
                     CELLS, DRUGS, (int*)(W + O_LAST), W + O_DIS, TOTAL, NCELLS, Bn);
  hipLaunchKernelGGL(build_h0_k, ew((long)TOTAL * H), dim3(256), 0, stream,
                     W + O_PROJ, W + O_HD, (const int*)(W + O_LAST), W + O_H0, TOTAL, NCELLS, H);
  gemm(W + O_HW, W + O_H0, GCN_W, TOTAL, 128, 128, nullptr, nullptr, nullptr, 0);
  hipLaunchKernelGGL(gcn_agg_k, ew((long)TOTAL * H), dim3(256), 0, stream,
                     W + O_HW, W + O_DIS, CELLS, DRUGS, GCN_B, PRELU, W + O_H1, TOTAL, NCELLS, Bn, H);

  // ---- attentive readout ----
  hipLaunchKernelGGL(concat2_k, ew((long)TOTAL * 2 * H), dim3(256), 0, stream,
                     W + O_H0, W + O_H1, W + O_CAT, TOTAL, H);
  gemm(W + O_SB, W + O_CAT, RD_W1, TOTAL, 128, 2 * H, RD_B1, nullptr, nullptr, 1);
  hipLaunchKernelGGL(dot_rows_k, ew((long)TOTAL), dim3(256), 0, stream,
                     W + O_SB, RD_W2, RD_B2, W + O_SC, TOTAL, H, 0);
  hipLaunchKernelGGL(softmax_k, dim3(1), dim3(256), 0, stream, W + O_SC, OUT_ATTN, TOTAL);
  hipLaunchKernelGGL(weighted_sum_k, dim3(1), dim3(128), 0, stream, W + O_H1, OUT_ATTN, OUT_TOD, TOTAL, H);

  // ---- bilinear discriminator + pair prediction ----
  hipLaunchKernelGGL(dot_rows_k, dim3(1), dim3(128), 0, stream, DISCW, OUT_TOD, nullptr, W + O_V, H, H, 0);
  hipLaunchKernelGGL(dot_rows_k, ew((long)TOTAL), dim3(256), 0, stream,
                     W + O_H1, W + O_V, nullptr, OUT_DISC, TOTAL, H, 1);
  hipLaunchKernelGGL(pred_k, ew((long)Bn), dim3(256), 0, stream,
                     W + O_H1, CELLS, DRUGS, PRED_W, PRED_B, OUT_PRED, Bn, NCELLS, H);
}
